// TIGNN_17910013624371
// MI455X (gfx1250) — matrix operand order repaired and checked by reading, not compiled, not run on hardware
//
#include <hip/hip_runtime.h>
#include <math.h>

// ---------------------------------------------------------------------------
// Types for CDNA5 WMMA (wave32): bf16 operands, f32 accumulation.
// ---------------------------------------------------------------------------
typedef __bf16 bf16;
typedef __attribute__((ext_vector_type(16))) __bf16 v16bf;
typedef __attribute__((ext_vector_type(8)))  __bf16 v8bf;
typedef __attribute__((ext_vector_type(8)))  float  v8f;

static __device__ __forceinline__ float sigf(float x) { return 1.0f / (1.0f + __expf(-x)); }
static __device__ __forceinline__ float silu_bwd(float h) {
  float s = sigf(h);
  return s * (1.0f + h * (1.0f - s));
}

// ---------------------------------------------------------------------------
// Elementwise helpers
// ---------------------------------------------------------------------------
__global__ void k_f32_to_bf16(const float* __restrict__ s, bf16* __restrict__ d, size_t n) {
  size_t i = (size_t)blockIdx.x * blockDim.x + threadIdx.x;
  if (i < n) d[i] = (bf16)s[i];
}

// Store W^T zero-padded along K: wt[c*Kpad + r] = (r<R) ? w[r*C + c] : 0
// (w is [R,C] row-major; wt is [C,Kpad] row-major, Kpad >= R, Kpad % 32 == 0)
__global__ void k_f32_to_bf16_T(const float* __restrict__ w, bf16* __restrict__ wt,
                                int R, int C, int Kpad) {
  size_t i = (size_t)blockIdx.x * blockDim.x + threadIdx.x;
  size_t n = (size_t)C * Kpad;
  if (i >= n) return;
  int c = (int)(i / Kpad), r = (int)(i % Kpad);
  wt[i] = (r < R) ? (bf16)w[(size_t)r * C + c] : (bf16)0.0f;
}

__global__ void k_zero_f32(float* __restrict__ d, size_t n) {
  size_t i = (size_t)blockIdx.x * blockDim.x + threadIdx.x;
  if (i < n) d[i] = 0.0f;
}

__global__ void k_axpy(float* __restrict__ d, const float* __restrict__ s, size_t n) {
  size_t i = (size_t)blockIdx.x * blockDim.x + threadIdx.x;
  if (i < n) d[i] += s[i];
}

// ---------------------------------------------------------------------------
// Fast WMMA GEMM:  O[M,N] = act( A[M,K] @ B^T + bias ) [ * silu'(dmask) ]
// Requirements: K % 32 == 0, M % 16 == 0 (guaranteed by construction).
//   A: [M,K] row-major bf16
//   B: [N,K] row-major bf16 (output-column's K-vector contiguous)
// Block = 256 threads = 8 waves; each wave: 16 rows x 64 cols.
// B strip (64 cols x K) is staged once per block into LDS (shared by 8 waves),
// padded by 8 bf16/row so ds_read_b128 fragments are bank-conflict free.
// All four B fragments are fetched before the WMMA burst so the 8 ds_load_b128
// issue back-to-back and the 4 v_wmma run as a burst behind one drain wait.
// NFULL: N % 64 == 0 (no column guards anywhere).
// ---------------------------------------------------------------------------
template <bool NFULL>
__global__ __launch_bounds__(256) void k_gemm_fast(
    const bf16* __restrict__ A, const bf16* __restrict__ B,
    const float* __restrict__ bias,
    const bf16* __restrict__ dmask, int maskMod,
    bf16* __restrict__ obf, float* __restrict__ of, bf16* __restrict__ opre,
    int M, int K, int N, int act) {
  extern __shared__ char smem[];
  bf16* sB = (bf16*)smem;                 // [64][K + 8]
  const int ldS = K + 8;                  // padded row length (bf16 units)
  const int colBase0 = blockIdx.y * 64;

  // ---- cooperative stage of B strip into LDS (v8bf granules) ----
  {
    const int chunksPerCol = K >> 3;      // v8bf per column
    const int totalChunks = 64 * chunksPerCol;
    for (int idx = threadIdx.x; idx < totalChunks; idx += 256) {
      const int c = idx / chunksPerCol;
      const int kc = (idx - c * chunksPerCol) << 3;
      v8bf v;
      if (NFULL || (colBase0 + c) < N) {
        v = *(const v8bf*)(B + (size_t)(colBase0 + c) * K + kc);
      } else {
#pragma unroll
        for (int i = 0; i < 8; ++i) v[i] = (bf16)0.0f;
      }
      *(v8bf*)(sB + (size_t)c * ldS + kc) = v;
    }
  }
  __syncthreads();

  const int lane = threadIdx.x & 31;
  const int wave = threadIdx.x >> 5;
  const int rowBase = (blockIdx.x * 8 + wave) * 16;
  if (rowBase >= M) return;
  const int lr = lane & 15;
  const bool hi = lane >= 16;

  v8f acc[4];
#pragma unroll
  for (int t = 0; t < 4; ++t)
#pragma unroll
    for (int e = 0; e < 8; ++e) acc[t][e] = 0.0f;

  const bf16* Arow = A + (size_t)(rowBase + lr) * K;
  const int kAoff = hi ? 16 : 0;  // A frag: lanes<16 hold K 0-15; lanes>=16: 16-31
  const int kBoff = hi ? 16 : 0;  // B frag: lanes<16 hold K 0-15; lanes>=16: 16-31

  for (int k0 = 0; k0 < K; k0 += 32) {
    if (k0 + 96 < K) __builtin_prefetch(Arow + k0 + 96, 0, 1);
    v16bf a;
    {
      const v8bf lo  = *(const v8bf*)(Arow + k0 + kAoff);
      const v8bf hi8 = *(const v8bf*)(Arow + k0 + kAoff + 8);
#pragma unroll
      for (int i = 0; i < 8; ++i) { a[i] = lo[i]; a[8 + i] = hi8[i]; }
    }
    // Fetch ALL B fragments first (8x ds_load_b128 back-to-back), ...
    v16bf b[4];
#pragma unroll
    for (int t = 0; t < 4; ++t) {
      const bf16* Brow = sB + (size_t)(t * 16 + lr) * ldS;
      const v8bf b0 = *(const v8bf*)(Brow + k0 + kBoff);
      const v8bf b1 = *(const v8bf*)(Brow + k0 + kBoff + 8);
#pragma unroll
      for (int i = 0; i < 8; ++i) { b[t][i] = b0[i]; b[t][8 + i] = b1[i]; }
    }
    // ...then run the 4 WMMAs as a burst.
#pragma unroll
    for (int t = 0; t < 4; ++t)
      acc[t] = __builtin_amdgcn_wmma_f32_16x16x32_bf16(false, a, false, b[t], (short)0,
                                                       acc[t], false, false);
  }

  // Epilogue: D layout lane<16 -> col=lane, elem e -> row e ; lane>=16 -> row e+8
#pragma unroll
  for (int t = 0; t < 4; ++t) {
    const int c = colBase0 + t * 16 + lr;
    if (!NFULL && c >= N) continue;
    const float bv = bias ? bias[c] : 0.0f;
#pragma unroll
    for (int e = 0; e < 8; ++e) {
      const int row = rowBase + e + (hi ? 8 : 0);
      float v = acc[t][e] + bv;
      const size_t oidx = (size_t)row * N + c;
      if (opre) opre[oidx] = (bf16)v;
      if (act) { const float s = sigf(v); v *= s; }
      if (dmask) v *= silu_bwd((float)dmask[(size_t)(row % maskMod) * N + c]);
      if (obf) obf[oidx] = (bf16)v;
      if (of)  of[oidx]  = v;
    }
  }
}

// ---------------------------------------------------------------------------
// Graph/data-movement kernels
// ---------------------------------------------------------------------------
// node encoder input, zero-padded to K=32: [N,32] = [z(6) | n(2) | 0...]
__global__ void k_build_node_in(const float* __restrict__ z, const float* __restrict__ nv,
                                bf16* __restrict__ out, int N) {
  size_t i = (size_t)blockIdx.x * blockDim.x + threadIdx.x;
  if (i >= (size_t)N * 32) return;
  int n = (int)(i / 32), j = (int)(i % 32);
  float v = 0.0f;
  if (j < 6) v = z[(size_t)n * 6 + j];
  else if (j < 8) v = nv[(size_t)n * 2 + (j - 6)];
  out[i] = (bf16)v;
}

// edge encoder input, zero-padded to K=32: [E,32] = [u(3) | |u|(1) | 0...]
__global__ void k_build_edge_in(const float* __restrict__ q0, const long long* __restrict__ src,
                                const long long* __restrict__ dst, bf16* __restrict__ out, int E) {
  size_t e = (size_t)blockIdx.x * blockDim.x + threadIdx.x;
  if (e >= (size_t)E) return;
  const long long s = src[e], d = dst[e];
  float u0 = q0[s * 3 + 0] - q0[d * 3 + 0];
  float u1 = q0[s * 3 + 1] - q0[d * 3 + 1];
  float u2 = q0[s * 3 + 2] - q0[d * 3 + 2];
  float nm = sqrtf(u0 * u0 + u1 * u1 + u2 * u2);
  bf16* o = out + e * 32;
  o[0] = (bf16)u0; o[1] = (bf16)u1; o[2] = (bf16)u2; o[3] = (bf16)nm;
#pragma unroll
  for (int j = 4; j < 32; ++j) o[j] = (bf16)0.0f;
}

// edge concat: [E,384] = [ea | x[src] | x[dest]] (f32 -> bf16)
__global__ void k_build_edge_cat(const float* __restrict__ ea, const float* __restrict__ x,
                                 const long long* __restrict__ src, const long long* __restrict__ dst,
                                 bf16* __restrict__ out, int E, int N, int H) {
  size_t i = (size_t)blockIdx.x * blockDim.x + threadIdx.x;
  if (i >= (size_t)E * H) return;
  size_t e = i / H; int h = (int)(i % H);
  bf16* o = out + e * (size_t)(3 * H);
  o[h]         = (bf16)ea[i];
  o[H + h]     = (bf16)x[(size_t)src[e] * H + h];
  o[2 * H + h] = (bf16)x[(size_t)dst[e] * H + h];
}

__global__ void k_scatter_add(const float* __restrict__ v, const long long* __restrict__ dst,
                              float* __restrict__ agg, int E, int H) {
  size_t i = (size_t)blockIdx.x * blockDim.x + threadIdx.x;
  if (i >= (size_t)E * H) return;
  size_t e = i / H; int h = (int)(i % H);
  atomicAdd(&agg[(size_t)dst[e] * H + h], v[i]);
}

__global__ void k_build_node_cat(const float* __restrict__ x, const float* __restrict__ agg,
                                 bf16* __restrict__ out, int N, int H) {
  size_t i = (size_t)blockIdx.x * blockDim.x + threadIdx.x;
  if (i >= (size_t)N * H) return;
  size_t n = i / H; int h = (int)(i % H);
  out[n * (size_t)(2 * H) + h]     = (bf16)x[i];
  out[n * (size_t)(2 * H) + H + h] = (bf16)agg[i];
}

// decoder-head backward seed: g_dh2[c,n,h] = W2[h] * silu'(pre2[n,h]), channels stacked [2N,H]
__global__ void k_dec_bwd_init(const float* __restrict__ W2E, const bf16* __restrict__ preE2,
                               const float* __restrict__ W2S, const bf16* __restrict__ preS2,
                               bf16* __restrict__ g, int N, int H) {
  size_t i = (size_t)blockIdx.x * blockDim.x + threadIdx.x;
  size_t tot = (size_t)2 * N * H;
  if (i >= tot) return;
  int c = (i >= (size_t)N * H) ? 1 : 0;
  size_t rem = i - (size_t)c * N * H;
  int h = (int)(rem % H);
  float w = c ? W2S[h] : W2E[h];
  float p = c ? (float)preS2[rem] : (float)preE2[rem];
  g[i] = (bf16)(w * silu_bwd(p));
}

// gx[r,h] += g_nin[r, 0:H]   (g_nin stride 2H)
__global__ void k_gx_add_nin(float* __restrict__ gx, const float* __restrict__ gnin,
                             size_t rows, int H) {
  size_t i = (size_t)blockIdx.x * blockDim.x + threadIdx.x;
  if (i >= rows * (size_t)H) return;
  size_t r = i / H; int h = (int)(i % H);
  gx[i] += gnin[r * (size_t)(2 * H) + h];
}

// g_ea_res[c,e,h] = gea[c,e,h] + g_nin[c, dest[e], H + h]   -> bf16 (GEMM input)
__global__ void k_gather_gagg(const float* __restrict__ gnin, const float* __restrict__ gea,
                              const long long* __restrict__ dst, bf16* __restrict__ out,
                              int E, int N, int H) {
  size_t i = (size_t)blockIdx.x * blockDim.x + threadIdx.x;
  size_t tot = (size_t)2 * E * H;
  if (i >= tot) return;
  int h = (int)(i % H);
  size_t r = i / H;
  int c = (r >= (size_t)E) ? 1 : 0;
  size_t e = r - (size_t)c * E;
  float v = gea[i] + gnin[((size_t)c * N + (size_t)dst[e]) * (size_t)(2 * H) + H + h];
  out[i] = (bf16)v;
}

// gea[r,h] += g_ein[r, 0:H]   (g_ein stride 3H)
__global__ void k_gea_update(float* __restrict__ gea, const float* __restrict__ gein,
                             size_t rows, int H) {
  size_t i = (size_t)blockIdx.x * blockDim.x + threadIdx.x;
  if (i >= rows * (size_t)H) return;
  size_t r = i / H; int h = (int)(i % H);
  gea[i] += gein[r * (size_t)(3 * H) + h];
}

// scatter grads of x[src], x[dest] from g_ein[:, H:2H] / [:, 2H:3H]
__global__ void k_scatter_gx(const float* __restrict__ gein, const long long* __restrict__ src,
                             const long long* __restrict__ dst, float* __restrict__ gx,
                             int E, int N, int H) {
  size_t i = (size_t)blockIdx.x * blockDim.x + threadIdx.x;
  size_t tot = (size_t)2 * E * H;
  if (i >= tot) return;
  int h = (int)(i % H);
  size_t r = i / H;
  int c = (r >= (size_t)E) ? 1 : 0;
  size_t e = r - (size_t)c * E;
  const float* row = gein + r * (size_t)(3 * H);
  atomicAdd(&gx[((size_t)c * N + (size_t)src[e]) * H + h], row[H + h]);
  atomicAdd(&gx[((size_t)c * N + (size_t)dst[e]) * H + h], row[2 * H + h]);
}

// Assemble outputs: L(skew), M=Mlow*Mlow^T, dEdz, dSdz, E, S  -> d_out (86*N floats)
__global__ void k_outputs(const float* __restrict__ l, const float* __restrict__ m,
                          const float* __restrict__ Eo, const float* __restrict__ So,
                          const float* __restrict__ g8, float* __restrict__ out, int N) {
  int n = blockIdx.x * blockDim.x + threadIdx.x;
  if (n >= N) return;
  float Lm[6][6], Mm[6][6];
  for (int i = 0; i < 6; ++i)
    for (int j = 0; j < 6; ++j) { Lm[i][j] = 0.f; Mm[i][j] = 0.f; }
  int idx = 0;
  for (int i = 1; i < 6; ++i)
    for (int j = 0; j < i; ++j) Lm[i][j] = l[(size_t)n * 15 + idx++];
  idx = 0;
  for (int i = 0; i < 6; ++i)
    for (int j = 0; j <= i; ++j) Mm[i][j] = m[(size_t)n * 21 + idx++];
  float* oL = out + (size_t)n * 36;
  float* oM = out + (size_t)N * 36 + (size_t)n * 36;
  for (int i = 0; i < 6; ++i)
    for (int j = 0; j < 6; ++j) {
      oL[i * 6 + j] = Lm[i][j] - Lm[j][i];
      float s = 0.f;
      for (int k = 0; k < 6; ++k) s += Mm[i][k] * Mm[j][k];
      oM[i * 6 + j] = s;
    }
  for (int d = 0; d < 6; ++d) {
    out[(size_t)N * 72 + (size_t)n * 6 + d] = g8[(size_t)n * 8 + d];
    out[(size_t)N * 78 + (size_t)n * 6 + d] = g8[((size_t)N + n) * 8 + d];
  }
  out[(size_t)N * 84 + n] = Eo[n];
  out[(size_t)N * 85 + n] = So[n];
}

// ---------------------------------------------------------------------------
// Host orchestration
// ---------------------------------------------------------------------------
extern "C" void kernel_launch(void* const* d_in, const int* in_sizes, int n_in,
                              void* d_out, int out_size, void* d_ws, size_t ws_size,
                              hipStream_t stream) {
  (void)n_in; (void)out_size; (void)ws_size;
  const int H = 128;
  const int N = in_sizes[0] / 6;  // z: [N,6]
  const int E = in_sizes[3] / 2;  // edge_index: [2,E] int64

  const float* z  = (const float*)d_in[0];
  const float* nv = (const float*)d_in[1];
  const float* q0 = (const float*)d_in[2];
  const long long* ei   = (const long long*)d_in[3];
  const long long* esrc = ei;
  const long long* edst = ei + E;

  // params in setup_inputs() insertion order:
  // enc_node(4..9), enc_edge(10..15), edge_mlps p0..p2 (16..33),
  // node_mlps p0..p2 (34..51), dec_E(52..57), dec_S(58..63), dec_L(64..69), dec_M(70..75)
  static const int mbase[12] = {4, 10, 16, 22, 28, 34, 40, 46, 52, 58, 64, 70};
  static const int mdims[12][4] = {
      {8, 128, 128, 128},   {4, 128, 128, 128},
      {384, 128, 128, 128}, {384, 128, 128, 128}, {384, 128, 128, 128},
      {256, 128, 128, 128}, {256, 128, 128, 128}, {256, 128, 128, 128},
      {128, 128, 128, 1},   {128, 128, 128, 1},
      {128, 128, 128, 15},  {128, 128, 128, 21}};

  char* wsb = (char*)d_ws;
  size_t off = 0;
  auto alloc = [&](size_t bytes) -> char* {
    off = (off + 255) & ~(size_t)255;
    char* p = wsb + off;
    off += bytes;
    return p;
  };
  auto ew = [](size_t n) { return dim3((unsigned)((n + 255) / 256)); };

  // bf16 weights:
  //   wT = W^T, K zero-padded to multiple of 32 (forward; K-contiguous per output column)
  //   wO = W row-major (backward dY @ W^T; K-contiguous automatically)
  bf16* wT[12][3]; bf16* wO[12][3]; const float* bia[12][3];
  int  wKp[12][3];
  for (int m = 0; m < 12; ++m)
    for (int k = 0; k < 3; ++k) {
      const int R = mdims[m][k], C = mdims[m][k + 1];
      const int Kp = (R + 31) & ~31;
      wKp[m][k] = Kp;
      const size_t cnt = (size_t)R * C;
      wT[m][k] = (bf16*)alloc((size_t)C * Kp * 2);
      wO[m][k] = (bf16*)alloc(cnt * 2);
      bia[m][k] = (const float*)d_in[mbase[m] + 2 * k + 1];
      const float* W = (const float*)d_in[mbase[m] + 2 * k];
      k_f32_to_bf16_T<<<ew((size_t)C * Kp), 256, 0, stream>>>(W, wT[m][k], R, C, Kp);
      k_f32_to_bf16<<<ew(cnt), 256, 0, stream>>>(W, wO[m][k], cnt);
    }

  // activations / scratch
  bf16* in32_bf = (bf16*)alloc((size_t)N * 32 * 2);  // node enc input, padded K=32
  bf16* e32_bf  = (bf16*)alloc((size_t)E * 32 * 2);  // edge enc input, padded K=32
  float* x_f   = (float*)alloc((size_t)N * H * 4);
  float* ea_f  = (float*)alloc((size_t)E * H * 4);
  bf16* penc1 = (bf16*)alloc((size_t)N * H * 2);
  bf16* penc2 = (bf16*)alloc((size_t)N * H * 2);
  bf16* preD[4];  // E1, E2, S1, S2
  for (int i = 0; i < 4; ++i) preD[i] = (bf16*)alloc((size_t)N * H * 2);
  bf16 *pe1[3], *pe2[3], *pn1[3], *pn2[3];
  for (int p = 0; p < 3; ++p) {
    pe1[p] = (bf16*)alloc((size_t)E * H * 2);
    pe2[p] = (bf16*)alloc((size_t)E * H * 2);
    pn1[p] = (bf16*)alloc((size_t)N * H * 2);
    pn2[p] = (bf16*)alloc((size_t)N * H * 2);
  }
  bf16* x_bf    = (bf16*)alloc((size_t)N * H * 2);
  bf16* ncat_bf = (bf16*)alloc((size_t)N * 2 * H * 2);
  bf16* tA_bf   = (bf16*)alloc((size_t)2 * E * H * 2);  // fwd hidden1 / bwd edge hidden
  bf16* tB_bf   = (bf16*)alloc((size_t)2 * E * H * 2);  // fwd hidden2 / bwd edge hidden
  bf16* gear_bf = (bf16*)alloc((size_t)2 * E * H * 2);
  bf16* gA_bf   = (bf16*)alloc((size_t)2 * N * H * 2);
  bf16* gB_bf   = (bf16*)alloc((size_t)2 * N * H * 2);
  bf16* gx_bf   = (bf16*)alloc((size_t)2 * N * H * 2);
  float* gx_f   = (float*)alloc((size_t)2 * N * H * 4);
  float* g8_f   = (float*)alloc((size_t)2 * N * 8 * 4);
  float* E_out  = (float*)alloc((size_t)N * 4);
  float* S_out  = (float*)alloc((size_t)N * 4);
  float* l_f    = (float*)alloc((size_t)N * 15 * 4);
  float* m_f    = (float*)alloc((size_t)N * 21 * 4);
  // phase-aliased big regions
  char* R1 = alloc((size_t)2 * E * 3 * H * 4);  // bwd g_ein f32 [2E,384] ; fwd ecat bf16 [E,384]
  float* gein_f  = (float*)R1;
  bf16*  ecat_bf = (bf16*)R1;
  char* R2 = alloc((size_t)2 * E * H * 4);      // bwd gea f32 [2E,H] ; fwd ea_res f32 [E,H]
  float* gea_f  = (float*)R2;
  float* ea_res = (float*)R2;
  char* R3 = alloc((size_t)2 * N * 2 * H * 4);  // bwd g_nin f32 [2N,256] ; fwd agg + xres
  float* gnin_f = (float*)R3;
  float* agg    = (float*)R3;
  float* xres   = (float*)(R3 + (size_t)N * H * 4);

  // GEMM dispatcher: every call has K%32==0 and M%16==0 -> LDS-staged fast path only.
  auto gemm = [&](const bf16* A, const bf16* B, const float* bias,
                  const bf16* dmask, int maskMod, bf16* obf, float* of, bf16* opre,
                  int M, int K, int Nc, int act) {
    dim3 grid((unsigned)((M + 127) / 128), (unsigned)((Nc + 63) / 64));
    const size_t shmem = (size_t)64 * (K + 8) * 2;
    if ((Nc & 63) == 0)
      k_gemm_fast<true><<<grid, 256, shmem, stream>>>(A, B, bias, dmask, maskMod,
                                                      obf, of, opre, M, K, Nc, act);
    else
      k_gemm_fast<false><<<grid, 256, shmem, stream>>>(A, B, bias, dmask, maskMod,
                                                       obf, of, opre, M, K, Nc, act);
  };

  // ------------------------- forward -------------------------
  k_build_node_in<<<ew((size_t)N * 32), 256, 0, stream>>>(z, nv, in32_bf, N);
  k_build_edge_in<<<ew((size_t)E), 256, 0, stream>>>(q0, esrc, edst, e32_bf, E);

  // enc_node (stash pre-activations for backward); layer0 runs with zero-padded K=32
  gemm(in32_bf, wT[0][0], bia[0][0], 0, 1, tA_bf, 0, penc1, N, wKp[0][0], H, 1);
  gemm(tA_bf,   wT[0][1], bia[0][1], 0, 1, tB_bf, 0, penc2, N, H, H, 1);
  gemm(tB_bf,   wT[0][2], bia[0][2], 0, 1, 0, x_f, 0, N, H, H, 0);
  // enc_edge (no grad wrt z flows into it)
  gemm(e32_bf, wT[1][0], bia[1][0], 0, 1, tA_bf, 0, 0, E, wKp[1][0], H, 1);
  gemm(tA_bf,  wT[1][1], bia[1][1], 0, 1, tB_bf, 0, 0, E, H, H, 1);
  gemm(tB_bf,  wT[1][2], bia[1][2], 0, 1, 0, ea_f, 0, E, H, H, 0);

  for (int p = 0; p < 3; ++p) {
    const int em = 2 + p, nm = 5 + p;
    k_build_edge_cat<<<ew((size_t)E * H), 256, 0, stream>>>(ea_f, x_f, esrc, edst, ecat_bf, E, N, H);
    gemm(ecat_bf, wT[em][0], bia[em][0], 0, 1, tA_bf, 0, pe1[p], E, 3 * H, H, 1);
    gemm(tA_bf,   wT[em][1], bia[em][1], 0, 1, tB_bf, 0, pe2[p], E, H, H, 1);
    gemm(tB_bf,   wT[em][2], bia[em][2], 0, 1, 0, ea_res, 0, E, H, H, 0);
    k_zero_f32<<<ew((size_t)N * H), 256, 0, stream>>>(agg, (size_t)N * H);
    k_scatter_add<<<ew((size_t)E * H), 256, 0, stream>>>(ea_res, edst, agg, E, H);
    k_build_node_cat<<<ew((size_t)N * H), 256, 0, stream>>>(x_f, agg, ncat_bf, N, H);
    gemm(ncat_bf, wT[nm][0], bia[nm][0], 0, 1, tA_bf, 0, pn1[p], N, 2 * H, H, 1);
    gemm(tA_bf,   wT[nm][1], bia[nm][1], 0, 1, tB_bf, 0, pn2[p], N, H, H, 1);
    gemm(tB_bf,   wT[nm][2], bia[nm][2], 0, 1, 0, xres, 0, N, H, H, 0);
    k_axpy<<<ew((size_t)N * H), 256, 0, stream>>>(x_f, xres, (size_t)N * H);
    k_axpy<<<ew((size_t)E * H), 256, 0, stream>>>(ea_f, ea_res, (size_t)E * H);
  }

  // ------------------------- decoders -------------------------
  k_f32_to_bf16<<<ew((size_t)N * H), 256, 0, stream>>>(x_f, x_bf, (size_t)N * H);
  gemm(x_bf,  wT[8][0],  bia[8][0],  0, 1, tA_bf, 0, preD[0], N, H, H, 1);
  gemm(tA_bf, wT[8][1],  bia[8][1],  0, 1, tB_bf, 0, preD[1], N, H, H, 1);
  gemm(tB_bf, wT[8][2],  bia[8][2],  0, 1, 0, E_out, 0, N, H, 1, 0);
  gemm(x_bf,  wT[9][0],  bia[9][0],  0, 1, tA_bf, 0, preD[2], N, H, H, 1);
  gemm(tA_bf, wT[9][1],  bia[9][1],  0, 1, tB_bf, 0, preD[3], N, H, H, 1);
  gemm(tB_bf, wT[9][2],  bia[9][2],  0, 1, 0, S_out, 0, N, H, 1, 0);
  gemm(x_bf,  wT[10][0], bia[10][0], 0, 1, tA_bf, 0, 0, N, H, H, 1);
  gemm(tA_bf, wT[10][1], bia[10][1], 0, 1, tB_bf, 0, 0, N, H, H, 1);
  gemm(tB_bf, wT[10][2], bia[10][2], 0, 1, 0, l_f, 0, N, H, 15, 0);
  gemm(x_bf,  wT[11][0], bia[11][0], 0, 1, tA_bf, 0, 0, N, H, H, 1);
  gemm(tA_bf, wT[11][1], bia[11][1], 0, 1, tB_bf, 0, 0, N, H, H, 1);
  gemm(tB_bf, wT[11][2], bia[11][2], 0, 1, 0, m_f, 0, N, H, 21, 0);

  // ------------------------- backward (dE/dz, dS/dz; 2 channels stacked) -------------------------
  const float* W2E = (const float*)d_in[mbase[8] + 4];
  const float* W2S = (const float*)d_in[mbase[9] + 4];
  k_dec_bwd_init<<<ew((size_t)2 * N * H), 256, 0, stream>>>(W2E, preD[1], W2S, preD[3], gA_bf, N, H);
  // gh1 = (gh2 @ W1^T) * silu'(pre1), per channel (different decoder weights)
  gemm(gA_bf,                  wO[8][1], 0, preD[0], N, gB_bf,                  0, 0, N, H, H, 0);
  gemm(gA_bf + (size_t)N * H,  wO[9][1], 0, preD[2], N, gB_bf + (size_t)N * H,  0, 0, N, H, H, 0);
  // gx = gh1 @ W0^T
  gemm(gB_bf,                  wO[8][0], 0, 0, 1, 0, gx_f,                  0, N, H, H, 0);
  gemm(gB_bf + (size_t)N * H,  wO[9][0], 0, 0, 1, 0, gx_f + (size_t)N * H,  0, N, H, H, 0);

  k_zero_f32<<<ew((size_t)2 * E * H), 256, 0, stream>>>(gea_f, (size_t)2 * E * H);

  for (int p = 2; p >= 0; --p) {
    const int em = 2 + p, nm = 5 + p;
    k_f32_to_bf16<<<ew((size_t)2 * N * H), 256, 0, stream>>>(gx_f, gx_bf, (size_t)2 * N * H);
    // node MLP backward: g_xres = gx (residual), shared weights -> batched [2N]
    gemm(gx_bf, wO[nm][2], 0, pn2[p], N, gA_bf, 0, 0, 2 * N, H, H, 0);
    gemm(gA_bf, wO[nm][1], 0, pn1[p], N, gB_bf, 0, 0, 2 * N, H, H, 0);
    gemm(gB_bf, wO[nm][0], 0, 0, 1, 0, gnin_f, 0, 2 * N, H, 2 * H, 0);
    k_gx_add_nin<<<ew((size_t)2 * N * H), 256, 0, stream>>>(gx_f, gnin_f, (size_t)2 * N, H);
    // segment_sum backward = gather; add residual grad on ea
    k_gather_gagg<<<ew((size_t)2 * E * H), 256, 0, stream>>>(gnin_f, gea_f, edst, gear_bf, E, N, H);
    // edge MLP backward, batched [2E]
    gemm(gear_bf, wO[em][2], 0, pe2[p], E, tA_bf, 0, 0, 2 * E, H, H, 0);
    gemm(tA_bf,   wO[em][1], 0, pe1[p], E, tB_bf, 0, 0, 2 * E, H, H, 0);
    gemm(tB_bf,   wO[em][0], 0, 0, 1, 0, gein_f, 0, 2 * E, H, 3 * H, 0);
    k_gea_update<<<ew((size_t)2 * E * H), 256, 0, stream>>>(gea_f, gein_f, (size_t)2 * E, H);
    k_scatter_gx<<<ew((size_t)2 * E * H), 256, 0, stream>>>(gein_f, esrc, edst, gx_f, E, N, H);
  }

  // encoder backward -> grad wrt cat(z, n); keep first 6 columns (z)
  k_f32_to_bf16<<<ew((size_t)2 * N * H), 256, 0, stream>>>(gx_f, gx_bf, (size_t)2 * N * H);
  gemm(gx_bf, wO[0][2], 0, penc2, N, gA_bf, 0, 0, 2 * N, H, H, 0);
  gemm(gA_bf, wO[0][1], 0, penc1, N, gB_bf, 0, 0, 2 * N, H, H, 0);
  gemm(gB_bf, wO[0][0], 0, 0, 1, 0, g8_f, 0, 2 * N, H, 8, 0);

  k_outputs<<<ew((size_t)N), 256, 0, stream>>>(l_f, m_f, E_out, S_out, g8_f, (float*)d_out, N);
}
